// VoxelformerCrossAttention_87282325390015
// MI455X (gfx1250) — compile-verified, matrix operand order checked by reading, not run on hardware
//
#include <hip/hip_runtime.h>
#include <hip/hip_bf16.h>

typedef __attribute__((ext_vector_type(16))) _Float16 v16h;
typedef __attribute__((ext_vector_type(8)))  _Float16 v8h;
typedef __attribute__((ext_vector_type(8)))  float    v8f;

namespace {
constexpr int kB = 2, kN = 4, kBN = 8;
constexpr int kQ = 8 * 32 * 32;     // 8192
constexpr int kC = 256;
constexpr int kH = 8, kD = 32, kL = 4, kP = 4;
constexpr int kNV = 9520;           // sum of level h*w
}

// ---------------- f32 -> f16 elementwise ----------------
__global__ void cvt_f16_kernel(const float* __restrict__ in,
                               _Float16* __restrict__ out, size_t n) {
  size_t i = (size_t)blockIdx.x * blockDim.x + threadIdx.x;
  if (i < n) out[i] = (_Float16)in[i];
}

// ---------------- W (K x N) f32 -> Wt (N x K) f16 ----------------
__global__ void transpose_cvt_kernel(const float* __restrict__ W,
                                     _Float16* __restrict__ Wt,
                                     int K, int Ncols) {
  int i = blockIdx.x * blockDim.x + threadIdx.x;
  if (i >= K * Ncols) return;
  int n = i % Ncols, k = i / Ncols;
  Wt[(size_t)n * K + k] = (_Float16)W[i];
}

// ---------------- WMMA GEMM: C[M,N] = A[M,K] * Bt[N,K]^T + bias[N] ----------
// One wave computes a 16x64 strip (4 N-tiles) so the A fragment is loaded
// once per K-step and reused by 4 back-to-back WMMAs (4x less A traffic).
// 4 waves per block stacked in M -> block covers 64x64.
// Requires M % 64 == 0, N % 64 == 0, K % 32 == 0 (true for this workload).
template <typename OutT>
__global__ void wmma_gemm_kernel(const _Float16* __restrict__ A,
                                 const _Float16* __restrict__ Bt,
                                 const float* __restrict__ bias,
                                 OutT* __restrict__ Cout,
                                 int M, int Ncols, int K) {
  const int lane  = threadIdx.x & 31;
  const int wave  = threadIdx.x >> 5;
  const int mtile = (blockIdx.x * 4 + wave) * 16;
  const int ntile = blockIdx.y * 64;
  if (mtile >= M) return;                 // wave-uniform: EXEC stays all-1s

  const int l16  = lane & 15;
  const int half = lane >> 4;             // 0 or 1
  const int kb   = half * 8;              // lane's K sub-offset per ISA layout

  const _Float16* ap = A + (size_t)(mtile + l16) * K + kb;
  const _Float16* bp0 = Bt + (size_t)(ntile + 0  + l16) * K + kb;
  const _Float16* bp1 = Bt + (size_t)(ntile + 16 + l16) * K + kb;
  const _Float16* bp2 = Bt + (size_t)(ntile + 32 + l16) * K + kb;
  const _Float16* bp3 = Bt + (size_t)(ntile + 48 + l16) * K + kb;

  v8f acc0 = {}, acc1 = {}, acc2 = {}, acc3 = {};
  for (int k0 = 0; k0 < K; k0 += 32) {
    v8h alo = *(const v8h*)(ap + k0);          // K = k0+kb .. +7   (16B)
    v8h ahi = *(const v8h*)(ap + k0 + 16);     // K = k0+16+kb .. +7
    v16h a = __builtin_shufflevector(alo, ahi, 0,1,2,3,4,5,6,7,8,9,10,11,12,13,14,15);

    v8h b0l = *(const v8h*)(bp0 + k0), b0h = *(const v8h*)(bp0 + k0 + 16);
    v8h b1l = *(const v8h*)(bp1 + k0), b1h = *(const v8h*)(bp1 + k0 + 16);
    v8h b2l = *(const v8h*)(bp2 + k0), b2h = *(const v8h*)(bp2 + k0 + 16);
    v8h b3l = *(const v8h*)(bp3 + k0), b3h = *(const v8h*)(bp3 + k0 + 16);
    v16h b0 = __builtin_shufflevector(b0l, b0h, 0,1,2,3,4,5,6,7,8,9,10,11,12,13,14,15);
    v16h b1 = __builtin_shufflevector(b1l, b1h, 0,1,2,3,4,5,6,7,8,9,10,11,12,13,14,15);
    v16h b2 = __builtin_shufflevector(b2l, b2h, 0,1,2,3,4,5,6,7,8,9,10,11,12,13,14,15);
    v16h b3 = __builtin_shufflevector(b3l, b3h, 0,1,2,3,4,5,6,7,8,9,10,11,12,13,14,15);

    acc0 = __builtin_amdgcn_wmma_f32_16x16x32_f16(false, a, false, b0, (short)0, acc0, false, false);
    acc1 = __builtin_amdgcn_wmma_f32_16x16x32_f16(false, a, false, b1, (short)0, acc1, false, false);
    acc2 = __builtin_amdgcn_wmma_f32_16x16x32_f16(false, a, false, b2, (short)0, acc2, false, false);
    acc3 = __builtin_amdgcn_wmma_f32_16x16x32_f16(false, a, false, b3, (short)0, acc3, false, false);
  }

  const v8f accs[4] = {acc0, acc1, acc2, acc3};
#pragma unroll
  for (int j = 0; j < 4; ++j) {
    const int col = ntile + j * 16 + l16;
    const float bval = bias[col];
#pragma unroll
    for (int r = 0; r < 8; ++r) {
      int m = mtile + r + 8 * half;            // C/D VGPR layout
      Cout[(size_t)m * Ncols + col] = (OutT)(accs[j][r] + bval);
    }
  }
}

// ---------------- softmax over 16 contiguous logits ----------------
__global__ void softmax16_kernel(float* __restrict__ att, int total) {
  int i = blockIdx.x * blockDim.x + threadIdx.x;
  if (i >= total) return;
  float* p = att + (size_t)i * 16;
  float mx = -1e30f;
#pragma unroll
  for (int j = 0; j < 16; ++j) mx = fmaxf(mx, p[j]);
  float e[16], s = 0.f;
#pragma unroll
  for (int j = 0; j < 16; ++j) { e[j] = __expf(p[j] - mx); s += e[j]; }
  float inv = 1.f / s;
#pragma unroll
  for (int j = 0; j < 16; ++j) p[j] = e[j] * inv;
}

// ---------------- deformable bilinear gather ----------------
// one wave per (bn, q, h); lane = d in [0,32)
__global__ void deform_sample_kernel(const _Float16* __restrict__ vval, // (BN,NV,C) f16
                                     const float* __restrict__ off,     // (B,Q,256)
                                     const float* __restrict__ aw,      // (B,Q,128)
                                     const float* __restrict__ refp,    // (BN,Q,2)
                                     _Float16* __restrict__ outh) {     // (BN,Q,C) f16
  const int lane = threadIdx.x & 31;
  const int gw = blockIdx.x * (blockDim.x >> 5) + (threadIdx.x >> 5);
  const int h  = gw & (kH - 1);
  const int q  = (gw >> 3) & (kQ - 1);
  const int bn = gw >> 16;                 // 3 (H) + 13 (Q) bits
  const int b  = bn >> 2;                  // N = 4

  const float rx = refp[((size_t)bn * kQ + q) * 2 + 0];
  const float ry = refp[((size_t)bn * kQ + q) * 2 + 1];
  const float* op = off + ((size_t)b * kQ + q) * 256 + h * 32; // (h,l,p,2)
  const float* ap = aw  + ((size_t)b * kQ + q) * 128 + h * 16; // (h,l,p)
  const _Float16* vb = vval + (size_t)bn * kNV * kC + h * kD + lane;

  const int levH[kL] = {64, 32, 16, 8};
  const int levW[kL] = {112, 56, 28, 14};
  const int levS[kL] = {0, 7168, 8960, 9408};

  float acc = 0.f;
#pragma unroll
  for (int l = 0; l < kL; ++l) {
    const int hh = levH[l], ww = levW[l], s0 = levS[l];
#pragma unroll
    for (int p = 0; p < kP; ++p) {
      const float ox = op[l * 8 + p * 2 + 0];
      const float oy = op[l * 8 + p * 2 + 1];
      const float a  = ap[l * 4 + p];
      const float x = rx * (float)ww + ox - 0.5f;  // (rx + ox/w)*w - 0.5
      const float y = ry * (float)hh + oy - 0.5f;
      const float x0f = floorf(x), y0f = floorf(y);
      const float lx = x - x0f, ly = y - y0f;
      const int x0 = (int)x0f, y0 = (int)y0f;
      const int x1 = x0 + 1, y1 = y0 + 1;
      const bool vx0 = (x0 >= 0) & (x0 < ww);
      const bool vx1 = (x1 >= 0) & (x1 < ww);
      const bool vy0 = (y0 >= 0) & (y0 < hh);
      const bool vy1 = (y1 >= 0) & (y1 < hh);
      float g00 = 0.f, g01 = 0.f, g10 = 0.f, g11 = 0.f;
      if (vy0 & vx0) g00 = (float)vb[(size_t)(s0 + y0 * ww + x0) * kC];
      if (vy0 & vx1) g01 = (float)vb[(size_t)(s0 + y0 * ww + x1) * kC];
      if (vy1 & vx0) g10 = (float)vb[(size_t)(s0 + y1 * ww + x0) * kC];
      if (vy1 & vx1) g11 = (float)vb[(size_t)(s0 + y1 * ww + x1) * kC];
      acc += a * ((1.f - ly) * ((1.f - lx) * g00 + lx * g01) +
                  ly * ((1.f - lx) * g10 + lx * g11));
    }
  }
  outh[((size_t)bn * kQ + q) * kC + h * kD + lane] = (_Float16)acc;
}

// ---------------- residual + mask + N-reduction + normalize ----------------
__global__ void finalize_kernel(const float* __restrict__ query, // (B,Q,C)
                                const float* __restrict__ attn,  // (BN,Q,C), incl. bo
                                const int* __restrict__ mask,    // (BN,Q)
                                float* __restrict__ out) {       // (B,Q,C)
  size_t i = (size_t)blockIdx.x * blockDim.x + threadIdx.x;
  if (i >= (size_t)kB * kQ * kC) return;
  const int c = (int)(i & (kC - 1));
  const size_t bq = i >> 8;
  const int q = (int)(bq & (kQ - 1));
  const int b = (int)(bq >> 13);
  const float qv = query[i];
  float sum = 0.f, cnt = 0.f;
#pragma unroll
  for (int n = 0; n < kN; ++n) {
    const int bn = b * kN + n;
    const float m = (float)mask[(size_t)bn * kQ + q];
    sum += m * (qv + attn[((size_t)bn * kQ + q) * kC + c]);
    cnt += m;
  }
  out[i] = sum / fmaxf(cnt, 1.f);
}

extern "C" void kernel_launch(void* const* d_in, const int* in_sizes, int n_in,
                              void* d_out, int out_size, void* d_ws, size_t ws_size,
                              hipStream_t stream) {
  (void)in_sizes; (void)n_in; (void)out_size; (void)ws_size;
  const float* query = (const float*)d_in[0];   // (B,Z,Y,X,C) == (B,Q,C)
  const float* value = (const float*)d_in[1];   // (BN,NV,C)
  const float* refp  = (const float*)d_in[2];   // (BN,Q,2)
  const int*   mask  = (const int*)d_in[3];     // (BN,Q)
  const float* Wv    = (const float*)d_in[6];
  const float* bv    = (const float*)d_in[7];
  const float* Woff  = (const float*)d_in[8];
  const float* boff  = (const float*)d_in[9];
  const float* Watt  = (const float*)d_in[10];
  const float* batt  = (const float*)d_in[11];
  const float* Wo    = (const float*)d_in[12];
  const float* bo    = (const float*)d_in[13];
  float* out = (float*)d_out;

  char* ws = (char*)d_ws;
  size_t cur = 0;
  auto alloc = [&](size_t bytes) {
    void* p = ws + cur;
    cur += (bytes + 255) & ~(size_t)255;
    return p;
  };

  _Float16* value_h = (_Float16*)alloc((size_t)kBN * kNV * kC * 2); // 39 MB
  _Float16* query_h = (_Float16*)alloc((size_t)kB * kQ * kC * 2);   // 8.4 MB
  _Float16* WvT     = (_Float16*)alloc((size_t)kC * kC * 2);
  _Float16* WoffT   = (_Float16*)alloc((size_t)kC * 256 * 2);
  _Float16* WattT   = (_Float16*)alloc((size_t)kC * 128 * 2);
  _Float16* WoT     = (_Float16*)alloc((size_t)kC * kC * 2);
  _Float16* v_h     = (_Float16*)alloc((size_t)kBN * kNV * kC * 2); // 39 MB
  float*    off_f   = (float*)alloc((size_t)kB * kQ * 256 * 4);     // 16.8 MB
  float*    att_f   = (float*)alloc((size_t)kB * kQ * 128 * 4);     // 8.4 MB
  _Float16* out_h   = (_Float16*)alloc((size_t)kBN * kQ * kC * 2);  // 33.6 MB
  float*    attn_f  = (float*)alloc((size_t)kBN * kQ * kC * 4);     // 67.1 MB

  // --- precision conversions ---
  {
    size_t n1 = (size_t)kBN * kNV * kC;
    cvt_f16_kernel<<<(unsigned)((n1 + 255) / 256), 256, 0, stream>>>(value, value_h, n1);
    size_t n2 = (size_t)kB * kQ * kC;
    cvt_f16_kernel<<<(unsigned)((n2 + 255) / 256), 256, 0, stream>>>(query, query_h, n2);
  }
  transpose_cvt_kernel<<<(kC * kC + 255) / 256, 256, 0, stream>>>(Wv, WvT, kC, kC);
  transpose_cvt_kernel<<<(kC * 256 + 255) / 256, 256, 0, stream>>>(Woff, WoffT, kC, 256);
  transpose_cvt_kernel<<<(kC * 128 + 255) / 256, 256, 0, stream>>>(Watt, WattT, kC, 128);
  transpose_cvt_kernel<<<(kC * kC + 255) / 256, 256, 0, stream>>>(Wo, WoT, kC, kC);

  // --- GEMM 1: v = value * Wv + bv (f16 out; halves gather traffic) ---
  {
    int M = kBN * kNV;                         // 76160 (mult. of 64)
    dim3 g(M / 64, kC / 64);
    wmma_gemm_kernel<_Float16><<<g, 128, 0, stream>>>(value_h, WvT, bv, v_h, M, kC, kC);
  }
  // --- GEMM 2: off = q * Woff + boff (per-B only: qf is broadcast over N) ---
  {
    int M = kB * kQ;                           // 16384
    dim3 g(M / 64, 256 / 64);
    wmma_gemm_kernel<float><<<g, 128, 0, stream>>>(query_h, WoffT, boff, off_f, M, 256, kC);
  }
  // --- GEMM 3: att logits = q * Watt + batt (per-B) ---
  {
    int M = kB * kQ;
    dim3 g(M / 64, 128 / 64);
    wmma_gemm_kernel<float><<<g, 128, 0, stream>>>(query_h, WattT, batt, att_f, M, 128, kC);
  }
  // --- softmax over L*P = 16 (in place) ---
  {
    int total = kB * kQ * kH;                  // 131072
    softmax16_kernel<<<(total + 255) / 256, 256, 0, stream>>>(att_f, total);
  }
  // --- deformable bilinear gather -> f16 (BN,Q,C) ---
  {
    int waves = kBN * kQ * kH;                 // 524288, 4 waves/block
    deform_sample_kernel<<<waves / 4, 128, 0, stream>>>(v_h, off_f, att_f, refp, out_h);
  }
  // --- GEMM 4: attn = out * Wo + bo ---
  {
    int M = kBN * kQ;                          // 65536
    dim3 g(M / 64, kC / 64);
    wmma_gemm_kernel<float><<<g, 128, 0, stream>>>(out_h, WoT, bo, attn_f, M, kC, kC);
  }
  // --- residual + mask + reduce over N + normalize ---
  {
    size_t total = (size_t)kB * kQ * kC;       // 4.19M
    finalize_kernel<<<(unsigned)((total + 255) / 256), 256, 0, stream>>>(query, attn_f, mask, out);
  }
}
